// GeodesicChunkPredictor_32177894982331
// MI455X (gfx1250) — matrix operand order repaired and checked by reading, not compiled, not run on hardware
//
#include <hip/hip_runtime.h>
#include <hip/hip_bf16.h>
#include <math.h>

// ---------- types ----------
typedef __attribute__((ext_vector_type(16))) __bf16 v16bf;
typedef __attribute__((ext_vector_type(8)))  float  v8f;
typedef __attribute__((ext_vector_type(4)))  float  v4f;

// ---------- problem constants ----------
#define BROWS   8192
#define HIDDIM  4096
#define HD      256
#define NFUSE   320      // 256 (W0) + 64 (Wg1)
#define MSTRIPS (BROWS/16)       // 512
#define KB1     (HIDDIM/32)      // 128 k-blocks for GEMM1
#define KB2     (HD/32)          // 8 k-blocks for GEMM2/3
#define NT1     (NFUSE/16)       // 20 n-tiles
#define NT2     (HD/16)          // 16 n-tiles

// ---------- workspace layout (bytes) ----------
static constexpr size_t OFF_BW  = 0;                                    // [W0|Wg1] frags bf16
static constexpr size_t OFF_B1  = OFF_BW + (size_t)HIDDIM*NFUSE*2;      // W1 frags
static constexpr size_t OFF_B2  = OFF_B1 + (size_t)HD*HD*2;             // W2 frags
static constexpr size_t OFF_Y   = OFF_B2 + (size_t)HD*HD*2;             // raw GEMM out f32 (8192x320 max)
static constexpr size_t OFF_X   = OFF_Y  + (size_t)BROWS*NFUSE*4;       // activation f32 8192x256
static constexpr size_t OFF_G   = OFF_X  + (size_t)BROWS*HD*4;          // gripper pre f32 8192x64
static constexpr size_t OFF_GEO = OFF_G  + (size_t)BROWS*64*4;          // anchors+rel-xi: 8192x66 f32

// =====================================================================
// Pack a row-major f32 weight (Kdim x Nsrc) into WMMA B-fragment layout.
// B frag (32x16 bf16): lane l holds col n0+(l&15); elem e -> K=kb+(l<16?0:16)+e.
// dst index = (kBlk*nTilesTot + nTile0 + nt)*32 + lane   (one-time pack)
// =====================================================================
__global__ __launch_bounds__(256) void pack_b_frag(const float* __restrict__ src,
                                                   v16bf* __restrict__ dst,
                                                   int KB, int nTilesTot, int nTile0,
                                                   int ntThis, int Nsrc) {
  int wid = threadIdx.x >> 5, lane = threadIdx.x & 31;
  int tile = blockIdx.x * 8 + wid;
  if (tile >= KB * ntThis) return;
  int kBlk = tile / ntThis, nt = tile % ntThis;
  int col = nt * 16 + (lane & 15);
  int kb  = kBlk * 32 + ((lane >> 4) << 4);
  v16bf o;
#pragma unroll
  for (int e = 0; e < 16; ++e)
    o[e] = (__bf16)src[(size_t)(kb + e) * Nsrc + col];
  dst[((size_t)kBlk * nTilesTot + nTile0 + nt) * 32 + lane] = o;
}

// =====================================================================
// Direct-A fragment GEMM: C(M x N, f32) = bf16(A_f32 row-major) @ B(bf16 frags)
// A fragments are built inline: 4x v4f contiguous loads per k-block per lane
// (lane l -> row m0+(l&15), K base (l<16?0:16): elems {b..b+15})
// + packed f32->bf16 cvts. No pack pass, no LDS; duplicate A reads across the
// wps waves of a strip hit L2 (operand is L2-resident at 192MB).
// B streams fragment-native 32B/lane b128 loads straight into v_wmma.
// =====================================================================
template <int NT>
__global__ __launch_bounds__(256) void gemm_direct(const float* __restrict__ A,
                                                   const v16bf* __restrict__ Bm,
                                                   float* __restrict__ C,
                                                   int KB, int nTilesTot, int wps,
                                                   int Kdim, int N, int Mstrips) {
  int wid = threadIdx.x >> 5, lane = threadIdx.x & 31;
  int gw = blockIdx.x * 8 + wid;
  if (gw >= Mstrips * wps) return;
  int mStrip = gw / wps;
  int ntile0 = (gw % wps) * NT;

  v8f zero = {0.f,0.f,0.f,0.f,0.f,0.f,0.f,0.f};
  v8f acc[NT];
#pragma unroll
  for (int t = 0; t < NT; ++t) acc[t] = zero;

  int m = mStrip * 16 + (lane & 15);
  const float* arow = A + (size_t)m * Kdim + ((lane >> 4) << 4);

  for (int kb = 0; kb < KB; ++kb) {
    const float* ap = arow + kb * 32;
    v4f a0 = *(const v4f*)(ap);
    v4f a1 = *(const v4f*)(ap + 4);
    v4f a2 = *(const v4f*)(ap + 8);
    v4f a3 = *(const v4f*)(ap + 12);
    v16bf a;
    a[0]=(__bf16)a0.x;  a[1]=(__bf16)a0.y;  a[2]=(__bf16)a0.z;  a[3]=(__bf16)a0.w;
    a[4]=(__bf16)a1.x;  a[5]=(__bf16)a1.y;  a[6]=(__bf16)a1.z;  a[7]=(__bf16)a1.w;
    a[8]=(__bf16)a2.x;  a[9]=(__bf16)a2.y;  a[10]=(__bf16)a2.z; a[11]=(__bf16)a2.w;
    a[12]=(__bf16)a3.x; a[13]=(__bf16)a3.y; a[14]=(__bf16)a3.z; a[15]=(__bf16)a3.w;

    const v16bf* Bp = Bm + ((size_t)kb * nTilesTot + ntile0) * 32 + lane;
#pragma unroll
    for (int t = 0; t < NT; ++t) {
      v16bf b = Bp[(size_t)t * 32];
      acc[t] = __builtin_amdgcn_wmma_f32_16x16x32_bf16(
          false, a, false, b, (short)0, acc[t], false, false);
    }
  }
  // C/D layout: lanes 0-15 -> N=lane, M=v ; lanes 16-31 -> N=lane-16, M=8+v
  int mbase = mStrip * 16 + ((lane >> 4) << 3);
#pragma unroll
  for (int t = 0; t < NT; ++t) {
    int col = (ntile0 + t) * 16 + (lane & 15);
#pragma unroll
    for (int v = 0; v < 8; ++v)
      C[(size_t)(mbase + v) * N + col] = acc[t][v];
  }
}

// ---------- activation helpers ----------
__device__ __forceinline__ float gelu_exact(float x) {
  return 0.5f * x * (1.f + erff(x * 0.7071067811865476f));
}
__device__ __forceinline__ float wave_sum(float s) {
#pragma unroll
  for (int m = 16; m >= 1; m >>= 1) s += __shfl_xor(s, m, 32);
  return s;
}

// GEMM1 epilogue: cols 0..255 -> bias+GELU+LN -> X ; cols 256..319 -> bias+GELU -> Gpre
__global__ __launch_bounds__(256) void postproc1(const float* __restrict__ Y,
                                                 const float* __restrict__ b0,
                                                 const float* __restrict__ g0,
                                                 const float* __restrict__ be0,
                                                 const float* __restrict__ bg1,
                                                 float* __restrict__ X,
                                                 float* __restrict__ Gp) {
  int wid = threadIdx.x >> 5, lane = threadIdx.x & 31;
  int r = blockIdx.x * 8 + wid;
  const float* yrow = Y + (size_t)r * NFUSE;
  int c0 = lane * 8;
  float vals[8];
  float s = 0.f;
#pragma unroll
  for (int i = 0; i < 8; ++i) {
    float x = gelu_exact(yrow[c0 + i] + b0[c0 + i]);
    vals[i] = x; s += x;
  }
  float mu = wave_sum(s) * (1.f / 256.f);
  float s2 = 0.f;
#pragma unroll
  for (int i = 0; i < 8; ++i) { float d = vals[i] - mu; s2 += d * d; }
  float inv = rsqrtf(wave_sum(s2) * (1.f / 256.f) + 1e-5f);
#pragma unroll
  for (int i = 0; i < 8; ++i)
    X[(size_t)r * HD + c0 + i] = (vals[i] - mu) * inv * g0[c0 + i] + be0[c0 + i];
#pragma unroll
  for (int i = 0; i < 2; ++i) {
    int c = lane * 2 + i;
    Gp[(size_t)r * 64 + c] = gelu_exact(yrow[256 + c] + bg1[c]);
  }
}

// GEMM2/3 epilogue: bias + GELU + LN (stride 256)
__global__ __launch_bounds__(256) void postproc_ln(const float* __restrict__ Y,
                                                   const float* __restrict__ bb,
                                                   const float* __restrict__ gg,
                                                   const float* __restrict__ be,
                                                   float* __restrict__ X) {
  int wid = threadIdx.x >> 5, lane = threadIdx.x & 31;
  int r = blockIdx.x * 8 + wid;
  const float* yrow = Y + (size_t)r * HD;
  int c0 = lane * 8;
  float vals[8];
  float s = 0.f;
#pragma unroll
  for (int i = 0; i < 8; ++i) {
    float x = gelu_exact(yrow[c0 + i] + bb[c0 + i]);
    vals[i] = x; s += x;
  }
  float mu = wave_sum(s) * (1.f / 256.f);
  float s2 = 0.f;
#pragma unroll
  for (int i = 0; i < 8; ++i) { float d = vals[i] - mu; s2 += d * d; }
  float inv = rsqrtf(wave_sum(s2) * (1.f / 256.f) + 1e-5f);
#pragma unroll
  for (int i = 0; i < 8; ++i)
    X[(size_t)r * HD + c0 + i] = (vals[i] - mu) * inv * gg[c0 + i] + be[c0 + i];
}

// ---------- SE(3) helpers ----------
__device__ __forceinline__ void m3mul(const float* A, const float* Bm, float* C) {
#pragma unroll
  for (int i = 0; i < 3; ++i)
#pragma unroll
    for (int j = 0; j < 3; ++j)
      C[i*3+j] = A[i*3+0]*Bm[0*3+j] + A[i*3+1]*Bm[1*3+j] + A[i*3+2]*Bm[2*3+j];
}

__device__ void exp_se3f(const float* xi, float* R, float* t) {
  float wx = xi[0], wy = xi[1], wz = xi[2];
  float th2 = wx*wx + wy*wy + wz*wz;
  bool sm = th2 < 1e-8f;
  float th2s = sm ? 1.f : th2;
  float th = sqrtf(th2s);
  float sn = sinf(th), cs = cosf(th);
  float A  = sm ? 1.f - th2/6.f        : sn/th;
  float Bc = sm ? 0.5f - th2/24.f      : (1.f - cs)/th2s;
  float Cc = sm ? (1.f/6.f - th2/120.f): (th - sn)/(th2s*th);
  float H[9]  = {0.f,-wz, wy,  wz,0.f,-wx,  -wy, wx,0.f};
  float H2[9]; m3mul(H, H, H2);
#pragma unroll
  for (int i = 0; i < 9; ++i) {
    float id = (i==0 || i==4 || i==8) ? 1.f : 0.f;
    R[i] = id + A*H[i] + Bc*H2[i];
  }
  float V[9];
#pragma unroll
  for (int i = 0; i < 9; ++i) {
    float id = (i==0 || i==4 || i==8) ? 1.f : 0.f;
    V[i] = id + Bc*H[i] + Cc*H2[i];
  }
#pragma unroll
  for (int i = 0; i < 3; ++i)
    t[i] = V[i*3]*xi[3] + V[i*3+1]*xi[4] + V[i*3+2]*xi[5];
}

__device__ void log_se3f(const float* R, const float* t, float* xi) {
  float tr = R[0] + R[4] + R[8];
  float cs = fminf(fmaxf((tr - 1.f)*0.5f, -1.f + 1e-7f), 1.f - 1e-7f);
  float th = acosf(cs);
  float th2 = th*th;
  bool sm = th < 1e-4f;
  float ths  = sm ? 1.f : th;
  float th2s = sm ? 1.f : th2;
  float sins = sinf(ths);
  float a = sm ? 0.5f*(1.f + th2/6.f) : ths/(2.f*sins);
  float wx = a*(R[7]-R[5]), wy = a*(R[2]-R[6]), wz = a*(R[3]-R[1]);
  float e = sm ? (1.f/12.f + th2/720.f) : (1.f/th2s - (1.f+cs)/(2.f*ths*sins));
  float H[9]  = {0.f,-wz, wy,  wz,0.f,-wx,  -wy, wx,0.f};
  float H2[9]; m3mul(H, H, H2);
  float Vi[9];
#pragma unroll
  for (int i = 0; i < 9; ++i) {
    float id = (i==0 || i==4 || i==8) ? 1.f : 0.f;
    Vi[i] = id - 0.5f*H[i] + e*H2[i];
  }
  xi[0] = wx; xi[1] = wy; xi[2] = wz;
#pragma unroll
  for (int i = 0; i < 3; ++i)
    xi[3+i] = Vi[i*3]*t[0] + Vi[i*3+1]*t[1] + Vi[i*3+2]*t[2];
}

// Per-row: Wp head -> 4 anchor exp_se3 -> 3 relative log_se3. 66 floats out/row.
__global__ __launch_bounds__(256) void geo_anchors(const float* __restrict__ X,
                                                   const float* __restrict__ Wp,
                                                   const float* __restrict__ bp,
                                                   float* __restrict__ geo) {
  int r = blockIdx.x * blockDim.x + threadIdx.x;
  if (r >= BROWS) return;
  float acc[24];
#pragma unroll
  for (int j = 0; j < 24; ++j) acc[j] = bp[j];
  const float* xr = X + (size_t)r * HD;
  for (int c = 0; c < HD; ++c) {
    float xv = xr[c];
    const float* wr = Wp + (size_t)c * 24;
#pragma unroll
    for (int j = 0; j < 24; ++j) acc[j] += xv * wr[j];
  }
  float anch[48];  // 4 x (R[9], t[3])
#pragma unroll
  for (int k = 0; k < 4; ++k) exp_se3f(&acc[k*6], &anch[k*12], &anch[k*12+9]);
  float* gr = geo + (size_t)r * 66;
#pragma unroll
  for (int i = 0; i < 48; ++i) gr[i] = anch[i];
#pragma unroll
  for (int k = 0; k < 3; ++k) {
    const float *R0 = &anch[k*12], *t0 = &anch[k*12+9];
    const float *R1 = &anch[(k+1)*12], *t1 = &anch[(k+1)*12+9];
    float Rr[9], tr3[3];
#pragma unroll
    for (int i = 0; i < 3; ++i)
#pragma unroll
      for (int j = 0; j < 3; ++j)
        Rr[i*3+j] = R0[0*3+i]*R1[0*3+j] + R0[1*3+i]*R1[1*3+j] + R0[2*3+i]*R1[2*3+j];
    float dt[3] = {t1[0]-t0[0], t1[1]-t0[1], t1[2]-t0[2]};
#pragma unroll
    for (int i = 0; i < 3; ++i)
      tr3[i] = R0[0*3+i]*dt[0] + R0[1*3+i]*dt[1] + R0[2*3+i]*dt[2];
    log_se3f(Rr, tr3, &gr[48 + k*6]);
  }
}

// Per (row, step): exp_se3(t*xi) composed with segment anchor -> 4x4 action.
__global__ __launch_bounds__(256) void geo_actions(const float* __restrict__ geo,
                                                   float* __restrict__ out) {
  int idx = blockIdx.x * blockDim.x + threadIdx.x;
  if (idx >= BROWS * 16) return;
  int r = idx >> 4, n = idx & 15;
  int k = (n < 5) ? 0 : ((n < 10) ? 1 : 2);
  int i = n - k * 5;
  float tv = (float)i / ((k == 2) ? 5.f : 4.f);
  const float* gr = geo + (size_t)r * 66;
  float R0[9], t0[3], xi[6];
#pragma unroll
  for (int j = 0; j < 9; ++j) R0[j] = gr[k*12 + j];
#pragma unroll
  for (int j = 0; j < 3; ++j) t0[j] = gr[k*12 + 9 + j];
#pragma unroll
  for (int j = 0; j < 6; ++j) xi[j] = tv * gr[48 + k*6 + j];
  float Rt[9], tt[3];
  exp_se3f(xi, Rt, tt);
  float Ro[9], to[3];
  m3mul(R0, Rt, Ro);
#pragma unroll
  for (int a = 0; a < 3; ++a)
    to[a] = R0[a*3]*tt[0] + R0[a*3+1]*tt[1] + R0[a*3+2]*tt[2] + t0[a];
  float* o = out + ((size_t)r * 16 + n) * 16;
#pragma unroll
  for (int a = 0; a < 3; ++a) {
#pragma unroll
    for (int b = 0; b < 3; ++b) o[a*4 + b] = Ro[a*3 + b];
    o[a*4 + 3] = to[a];
  }
  o[12] = 0.f; o[13] = 0.f; o[14] = 0.f; o[15] = 1.f;
}

// gripper = sigmoid(Gpre @ Wg2 + bg2)
__global__ __launch_bounds__(256) void gripper_head(const float* __restrict__ Gp,
                                                    const float* __restrict__ Wg2,
                                                    const float* __restrict__ bg2,
                                                    float* __restrict__ out) {
  int idx = blockIdx.x * blockDim.x + threadIdx.x;
  if (idx >= BROWS * 16) return;
  int r = idx >> 4, j = idx & 15;
  float s = bg2[j];
  const float* gr = Gp + (size_t)r * 64;
#pragma unroll
  for (int c = 0; c < 64; ++c) s += gr[c] * Wg2[c * 16 + j];
  out[(size_t)BROWS * 16 * 16 + (size_t)r * 16 + j] = 1.f / (1.f + expf(-s));
}

// =====================================================================
extern "C" void kernel_launch(void* const* d_in, const int* in_sizes, int n_in,
                              void* d_out, int out_size, void* d_ws, size_t ws_size,
                              hipStream_t stream) {
  const float* h   = (const float*)d_in[0];
  const float* W0  = (const float*)d_in[1];
  const float* b0  = (const float*)d_in[2];
  const float* g0  = (const float*)d_in[3];
  const float* be0 = (const float*)d_in[4];
  const float* W1  = (const float*)d_in[5];
  const float* b1  = (const float*)d_in[6];
  const float* g1  = (const float*)d_in[7];
  const float* be1 = (const float*)d_in[8];
  const float* W2  = (const float*)d_in[9];
  const float* b2  = (const float*)d_in[10];
  const float* g2  = (const float*)d_in[11];
  const float* be2 = (const float*)d_in[12];
  const float* Wp  = (const float*)d_in[13];
  const float* bp  = (const float*)d_in[14];
  const float* Wg1 = (const float*)d_in[15];
  const float* bg1 = (const float*)d_in[16];
  const float* Wg2 = (const float*)d_in[17];
  const float* bg2 = (const float*)d_in[18];
  (void)in_sizes; (void)n_in; (void)out_size; (void)ws_size;

  char* ws = (char*)d_ws;
  v16bf* Bw = (v16bf*)(ws + OFF_BW);
  v16bf* B1 = (v16bf*)(ws + OFF_B1);
  v16bf* B2 = (v16bf*)(ws + OFF_B2);
  float* Y  = (float*)(ws + OFF_Y);
  float* X  = (float*)(ws + OFF_X);
  float* Gp = (float*)(ws + OFF_G);
  float* geo = (float*)(ws + OFF_GEO);
  float* out = (float*)d_out;

  // one-time weight packs (KB-scale, L2-resident)
  pack_b_frag<<<KB1*16/8, 256, 0, stream>>>(W0,  Bw, KB1, NT1, 0,  16, HD);
  pack_b_frag<<<KB1*4/8,  256, 0, stream>>>(Wg1, Bw, KB1, NT1, 16, 4,  64);
  pack_b_frag<<<KB2*NT2/8, 256, 0, stream>>>(W1, B1, KB2, NT2, 0, NT2, HD);
  pack_b_frag<<<KB2*NT2/8, 256, 0, stream>>>(W2, B2, KB2, NT2, 0, NT2, HD);

  // fused GEMM1: [x_pre | gripper_pre] = h @ [W0 | Wg1]  (8192x4096x320, bf16 WMMA,
  // A converted inline from f32 -> single HBM pass over h)
  gemm_direct<10><<<MSTRIPS*2/8, 256, 0, stream>>>(h, Bw, Y, KB1, NT1, 2, HIDDIM, NFUSE, MSTRIPS);
  postproc1<<<BROWS/8, 256, 0, stream>>>(Y, b0, g0, be0, bg1, X, Gp);

  // layer 2
  gemm_direct<8><<<MSTRIPS*2/8, 256, 0, stream>>>(X, B1, Y, KB2, NT2, 2, HD, HD, MSTRIPS);
  postproc_ln<<<BROWS/8, 256, 0, stream>>>(Y, b1, g1, be1, X);

  // layer 3
  gemm_direct<8><<<MSTRIPS*2/8, 256, 0, stream>>>(X, B2, Y, KB2, NT2, 2, HD, HD, MSTRIPS);
  postproc_ln<<<BROWS/8, 256, 0, stream>>>(Y, b2, g2, be2, X);

  // SE(3) heads
  geo_anchors<<<BROWS/256, 256, 0, stream>>>(X, Wp, bp, geo);
  geo_actions<<<BROWS*16/256, 256, 0, stream>>>(geo, out);
  gripper_head<<<BROWS*16/256, 256, 0, stream>>>(Gp, Wg2, bg2, out);
}
